// LongTermMemory_42442866819863
// MI455X (gfx1250) — matrix operand order repaired and checked by reading, not compiled, not run on hardware
//
#include <hip/hip_runtime.h>
#include <hip/hip_bf16.h>
#include <math.h>

#define D_MODEL  1024
#define MEM_SIZE 131072
#define TOPK     64
#define D_STATE  16
#define D_CONV   4
#define D_INNER  2048
#define DT_RANK  64
#define XDBL     96          // DT_RANK + 2*D_STATE
#define L_SEQ    64
#define EPS_COS  1e-8f
#define EPS_LN   1e-5f

typedef __attribute__((ext_vector_type(16))) __bf16        v16bf;
typedef __attribute__((ext_vector_type(8)))  float         v8f;
typedef __attribute__((ext_vector_type(8)))  unsigned int  v8u;
typedef __attribute__((ext_vector_type(4)))  unsigned int  u32x4;
typedef __attribute__((ext_vector_type(8)))  int           i32x8;
typedef __attribute__((ext_vector_type(4)))  int           i32x4;

static __device__ __forceinline__ unsigned short f2bf(float f) {
    unsigned int u = __float_as_uint(f);
    unsigned int r = u + 0x7FFFu + ((u >> 16) & 1u);   // RNE
    return (unsigned short)(r >> 16);
}

static __device__ __forceinline__ float wave_red_add(float v) {
    #pragma unroll
    for (int off = 16; off > 0; off >>= 1) v += __shfl_down(v, off, 32);
    return v;
}

// ---- Tensor Data Mover: 2D tile (rows x rowlen, bf16) global -> LDS ----
// D# per CDNA5 ISA: group0 = {count|flags, lds_addr, ga_lo, ga_hi|type=2},
// group1 = mask/data_size/dims/tile/stride, groups 2-4 zero (2D tensor).
// This toolchain exposes the 6-arg builtin (amdgpu-toolchain / clang-23).
static __device__ __forceinline__ void tdm_load_tile_2d(const unsigned short* gptr,
                                                        unsigned lds_off,
                                                        unsigned rows, unsigned rowlen,
                                                        unsigned stride_elems) {
    unsigned long long ga = (unsigned long long)(const void*)gptr;
    u32x4 g0;
    g0[0] = 1u;                                                  // count=1, user desc
    g0[1] = lds_off;                                             // lds_addr (bytes)
    g0[2] = (unsigned)ga;                                        // global_addr[31:0]
    g0[3] = (unsigned)((ga >> 32) & 0x01FFFFFFu) | (2u << 30);   // addr[56:32], type=2
    i32x8 g1;
    g1[0] = 0x00010000;                            // wg_mask=0, data_size=1 (2 bytes)
    g1[1] = (int)((stride_elems & 0xFFFFu) << 16); // tensor_dim0[15:0] @ bits[79:48]
    g1[2] = (int)((rows & 0xFFFFu) << 16);         // tensor_dim1[15:0] @ bits[111:80]
    g1[3] = (int)((rowlen & 0xFFFFu) << 16);       // tile_dim0 @ bits[127:112]
    g1[4] = (int)(rows & 0xFFFFu);                 // tile_dim1 (tile_dim2 = 0)
    g1[5] = (int)stride_elems;                     // tensor_dim0_stride[31:0]
    g1[6] = 0;                                     // stride hi / dim1_stride lo
    g1[7] = 0;
    i32x4 z4 = {0, 0, 0, 0};
    i32x8 z8 = {0, 0, 0, 0, 0, 0, 0, 0};
    __builtin_amdgcn_tensor_load_to_lds(g0, g1, z4, z4, z8, 0);
}

// ---------------- stage 0: ||query|| ----------------
__global__ void qnorm_kernel(const float* __restrict__ q, float* __restrict__ qn) {
    __shared__ float red[256];
    float s = 0.f;
    for (int i = threadIdx.x; i < D_MODEL; i += 256) { float v = q[i]; s += v * v; }
    red[threadIdx.x] = s; __syncthreads();
    for (int st = 128; st > 0; st >>= 1) {
        if (threadIdx.x < st) red[threadIdx.x] += red[threadIdx.x + st];
        __syncthreads();
    }
    if (threadIdx.x == 0) qn[0] = sqrtf(red[0]);
}

// ---------------- stage 1: cosine sims (memory-bound, 512 MB stream) ----------------
__global__ void sims_kernel(const float* __restrict__ mem, const float* __restrict__ q,
                            const float* __restrict__ qn, float* __restrict__ sims) {
    const int lane = threadIdx.x & 31;
    const int wave = threadIdx.x >> 5;
    const int row  = blockIdx.x * 8 + wave;
    const float4* mrow = (const float4*)(mem + (size_t)row * D_MODEL);
    const float4* q4   = (const float4*)q;
    if (row + 8 < MEM_SIZE)   // stream-ahead hint -> global_prefetch_b8
        __builtin_prefetch(mem + (size_t)(row + 8) * D_MODEL + lane * 16, 0, 1);
    float dot = 0.f, sq = 0.f;
    #pragma unroll
    for (int j = 0; j < 8; ++j) {
        int i4 = j * 32 + lane;
        float4 m = mrow[i4];
        float4 qq = q4[i4];
        dot += m.x * qq.x + m.y * qq.y + m.z * qq.z + m.w * qq.w;
        sq  += m.x * m.x + m.y * m.y + m.z * m.z + m.w * m.w;
    }
    dot = wave_red_add(dot);
    sq  = wave_red_add(sq);
    if (lane == 0) {
        float denom = fmaxf(sqrtf(sq) * qn[0], EPS_COS);
        sims[row] = dot / denom;
    }
}

// ---------------- stage 2: copy usage to output ----------------
__global__ void copy_usage_kernel(const int* __restrict__ src, int* __restrict__ dst) {
    int i = blockIdx.x * 256 + threadIdx.x;
    if (i < MEM_SIZE) dst[i] = src[i];
}

// ---------------- stage 3: iterative top-64 (single workgroup) ----------------
__global__ void topk_kernel(float* __restrict__ sims, int* __restrict__ idx_out) {
    __shared__ float vals[1024];
    __shared__ int   idxs[1024];
    const int tid = threadIdx.x;
    for (int t = 0; t < TOPK; ++t) {
        float best = -INFINITY; int bi = tid;
        for (int i = tid; i < MEM_SIZE; i += 1024) {
            float v = sims[i];
            if (v > best) { best = v; bi = i; }
        }
        vals[tid] = best; idxs[tid] = bi;
        __syncthreads();
        for (int st = 512; st > 0; st >>= 1) {
            if (tid < st && vals[tid + st] > vals[tid]) {
                vals[tid] = vals[tid + st]; idxs[tid] = idxs[tid + st];
            }
            __syncthreads();
        }
        if (tid == 0) { idx_out[t] = idxs[0]; sims[idxs[0]] = -INFINITY; }
        __syncthreads();
    }
}

// ---------------- stage 4: gather retrieved rows -> bf16, bump usage ----------------
__global__ void gather_kernel(const float* __restrict__ mem, const int* __restrict__ idx,
                              unsigned short* __restrict__ ret_bf, int* __restrict__ usage) {
    const int r = blockIdx.x;
    const int row = idx[r];
    const float* src = mem + (size_t)row * D_MODEL;
    for (int c = threadIdx.x; c < D_MODEL; c += 256)
        ret_bf[r * D_MODEL + c] = f2bf(src[c]);
    if (threadIdx.x == 0) atomicAdd(&usage[row], 1);
}

// ---------------- generic f32 -> bf16 convert ----------------
__global__ void cvt_bf16_kernel(const float* __restrict__ src, unsigned short* __restrict__ dst, int n) {
    int i = blockIdx.x * 256 + threadIdx.x;
    if (i < n) dst[i] = f2bf(src[i]);
}

// ---------------- WMMA bf16 GEMM (direct global loads) ----------------
// C[M,N] = A[M,K] * W[N,K]^T ; one wave per 16x16 tile, K step 32
__global__ void gemm_bf16_kernel(const unsigned short* __restrict__ A, int lda,
                                 const unsigned short* __restrict__ W, int ldw,
                                 float* __restrict__ C, int ldc, int K) {
    const int lane = threadIdx.x & 31;
    const int m    = lane & 15;
    const int kg   = lane >> 4;
    const unsigned short* Arow = A + (size_t)(blockIdx.y * 16 + m) * lda;
    const unsigned short* Wrow = W + (size_t)(blockIdx.x * 16 + m) * ldw;

    v8f c = {};
    for (int k0 = 0; k0 < K; k0 += 32) {
        v8u au;
        #pragma unroll
        for (int i = 0; i < 8; ++i) {
            int k = k0 + (i >> 2) * 16 + kg * 8 + (i & 3) * 2;
            au[i] = *(const unsigned int*)(Arow + k);
        }
        v8u bu;
        #pragma unroll
        for (int v = 0; v < 8; ++v) {
            int k = k0 + (v >> 2) * 16 + kg * 8 + (v & 3) * 2;
            bu[v] = *(const unsigned int*)(Wrow + k);
        }
        v16bf a = __builtin_bit_cast(v16bf, au);
        v16bf b = __builtin_bit_cast(v16bf, bu);
        c = __builtin_amdgcn_wmma_f32_16x16x32_bf16(false, a, false, b, (short)0, c, false, false);
    }
    #pragma unroll
    for (int r = 0; r < 8; ++r)
        C[(size_t)(blockIdx.y * 16 + kg * 8 + r) * ldc + blockIdx.x * 16 + m] = c[r];
}

// ---------------- WMMA bf16 GEMM (TDM double-buffered through LDS) ----------------
// Tensor Data Mover stages 16x32 A/B tiles into LDS (TENSORcnt-tracked),
// wave overlaps next tile's DMA with WMMA on current; fragments via ds_load_b128.
__global__ void gemm_bf16_tdm_kernel(const unsigned short* __restrict__ A, int lda,
                                     const unsigned short* __restrict__ W, int ldw,
                                     float* __restrict__ C, int ldc, int K) {
    __shared__ __align__(16) unsigned short Ab[2][16 * 32];
    __shared__ __align__(16) unsigned short Bb[2][16 * 32];
    const int lane = threadIdx.x & 31;
    const int m    = lane & 15;
    const int kg   = lane >> 4;
    const unsigned short* Atile = A + (size_t)(blockIdx.y * 16) * lda;
    const unsigned short* Wtile = W + (size_t)(blockIdx.x * 16) * ldw;

    const unsigned ldsA0 = (unsigned)(unsigned long long)(const void*)&Ab[0][0];
    const unsigned ldsA1 = (unsigned)(unsigned long long)(const void*)&Ab[1][0];
    const unsigned ldsB0 = (unsigned)(unsigned long long)(const void*)&Bb[0][0];
    const unsigned ldsB1 = (unsigned)(unsigned long long)(const void*)&Bb[1][0];

    // prime buffer 0
    tdm_load_tile_2d(Atile, ldsA0, 16, 32, (unsigned)lda);
    tdm_load_tile_2d(Wtile, ldsB0, 16, 32, (unsigned)ldw);

    v8f c = {};
    int it = 0;
    for (int k0 = 0; k0 < K; k0 += 32, ++it) {
        const int buf = it & 1;
        if (k0 + 32 < K) {    // kick next tile pair into the other buffer
            tdm_load_tile_2d(Atile + (k0 + 32), buf ? ldsA0 : ldsA1, 16, 32, (unsigned)lda);
            tdm_load_tile_2d(Wtile + (k0 + 32), buf ? ldsB0 : ldsB1, 16, 32, (unsigned)ldw);
            __builtin_amdgcn_s_wait_tensorcnt(2);   // current pair landed
        } else {
            __builtin_amdgcn_s_wait_tensorcnt(0);   // drain
        }
        // TDM wrote LDS behind the compiler's back (integer lds_addr in D#):
        // force it to treat LDS as clobbered before reading fragments.
        asm volatile("" ::: "memory");

        const unsigned short* ab = &Ab[buf][m * 32 + kg * 8];
        uint4 aLo = *(const uint4*)(ab);
        uint4 aHi = *(const uint4*)(ab + 16);
        const unsigned short* bb = &Bb[buf][m * 32 + kg * 8];
        uint4 bLo = *(const uint4*)(bb);
        uint4 bHi = *(const uint4*)(bb + 16);

        v8u au = {aLo.x, aLo.y, aLo.z, aLo.w, aHi.x, aHi.y, aHi.z, aHi.w};
        v8u bu = {bLo.x, bLo.y, bLo.z, bLo.w, bHi.x, bHi.y, bHi.z, bHi.w};
        v16bf a = __builtin_bit_cast(v16bf, au);
        v16bf b = __builtin_bit_cast(v16bf, bu);
        c = __builtin_amdgcn_wmma_f32_16x16x32_bf16(false, a, false, b, (short)0, c, false, false);
    }
    #pragma unroll
    for (int r = 0; r < 8; ++r)
        C[(size_t)(blockIdx.y * 16 + kg * 8 + r) * ldc + blockIdx.x * 16 + m] = c[r];
}

// ---------------- stage 6: depthwise conv (D_CONV=4) + SiLU ----------------
__global__ void conv_silu_kernel(const float* __restrict__ xz,
                                 const float* __restrict__ conv_w, const float* __restrict__ conv_b,
                                 float* __restrict__ u_f32, unsigned short* __restrict__ u_bf) {
    int t = blockIdx.x * 256 + threadIdx.x;
    if (t >= L_SEQ * D_INNER) return;
    int l = t / D_INNER, d = t % D_INNER;
    float acc = conv_b[d];
    #pragma unroll
    for (int j = 0; j < D_CONV; ++j) {
        int ls = l + j - (D_CONV - 1);
        if (ls >= 0) acc += xz[(size_t)ls * (2 * D_INNER) + d] * conv_w[d * D_CONV + j];
    }
    float s = acc / (1.f + expf(-acc));       // silu
    u_f32[t] = s;
    u_bf[t]  = f2bf(s);
}

// ---------------- stage 8: extract dt (x_dbl[:, :64]) -> bf16 ----------------
__global__ void dt_cvt_kernel(const float* __restrict__ x_dbl, unsigned short* __restrict__ dt_bf) {
    int t = blockIdx.x * 256 + threadIdx.x;
    if (t >= L_SEQ * DT_RANK) return;
    int l = t / DT_RANK, r = t % DT_RANK;
    dt_bf[t] = f2bf(x_dbl[l * XDBL + r]);
}

// ---------------- stage 10: selective scan; only y at l=63 is needed ----------------
__global__ void scan_kernel(const float* __restrict__ deltapre, const float* __restrict__ dt_b,
                            const float* __restrict__ x_dbl,    const float* __restrict__ u_f32,
                            const float* __restrict__ A_log,    const float* __restrict__ Dvec,
                            const float* __restrict__ xz,       float* __restrict__ yfinal) {
    __shared__ float bc[L_SEQ][2 * D_STATE];   // B (s<16) and C (s>=16) per step
    for (int t = threadIdx.x; t < L_SEQ * 2 * D_STATE; t += 256) {
        int l = t / (2 * D_STATE), s = t % (2 * D_STATE);
        bc[l][s] = x_dbl[l * XDBL + DT_RANK + s];
    }
    __syncthreads();

    const int d = blockIdx.x * 256 + threadIdx.x;   // 0 .. D_INNER-1
    float A[D_STATE], h[D_STATE];
    #pragma unroll
    for (int s = 0; s < D_STATE; ++s) { A[s] = -expf(A_log[d * D_STATE + s]); h[s] = 0.f; }
    const float db = dt_b[d];

    float ulast = 0.f;
    for (int l = 0; l < L_SEQ; ++l) {
        float dp = deltapre[(size_t)l * D_INNER + d] + db;
        float delta = (dp > 20.f) ? dp : log1pf(expf(dp));   // softplus
        float ul = u_f32[(size_t)l * D_INNER + d];
        ulast = ul;
        float dbu = delta * ul;
        #pragma unroll
        for (int s = 0; s < D_STATE; ++s)
            h[s] = expf(delta * A[s]) * h[s] + dbu * bc[l][s];
    }
    float y = 0.f;
    #pragma unroll
    for (int s = 0; s < D_STATE; ++s) y += h[s] * bc[L_SEQ - 1][D_STATE + s];
    y += ulast * Dvec[d];
    float z = xz[(size_t)(L_SEQ - 1) * (2 * D_INNER) + D_INNER + d];
    y *= z / (1.f + expf(-z));                // * silu(z)
    yfinal[d] = y;
}

// ---------------- stage 11: out_proj GEMV (1024 x 2048) ----------------
__global__ void outproj_gemv_kernel(const float* __restrict__ Wout, const float* __restrict__ y,
                                    float* __restrict__ ctx) {
    const int lane = threadIdx.x & 31;
    const int wave = threadIdx.x >> 5;
    const int e    = blockIdx.x * 8 + wave;
    const float4* wr = (const float4*)(Wout + (size_t)e * D_INNER);
    const float4* y4 = (const float4*)y;
    float acc = 0.f;
    #pragma unroll
    for (int j = 0; j < 16; ++j) {
        int i4 = j * 32 + lane;
        float4 w = wr[i4]; float4 v = y4[i4];
        acc += w.x * v.x + w.y * v.y + w.z * v.z + w.w * v.w;
    }
    acc = wave_red_add(acc);
    if (lane == 0) ctx[e] = acc;
}

// ---------------- stage 12: LayerNorm ----------------
__global__ void layernorm_kernel(const float* __restrict__ ctx, const float* __restrict__ g,
                                 const float* __restrict__ b, float* __restrict__ out) {
    __shared__ float s1[1024], s2[1024];
    const int i = threadIdx.x;
    float v = ctx[i];
    s1[i] = v; s2[i] = v * v;
    __syncthreads();
    for (int st = 512; st > 0; st >>= 1) {
        if (i < st) { s1[i] += s1[i + st]; s2[i] += s2[i + st]; }
        __syncthreads();
    }
    float mu  = s1[0] / D_MODEL;
    float var = s2[0] / D_MODEL - mu * mu;
    out[i] = (v - mu) * rsqrtf(var + EPS_LN) * g[i] + b[i];
}

extern "C" void kernel_launch(void* const* d_in, const int* in_sizes, int n_in,
                              void* d_out, int out_size, void* d_ws, size_t ws_size,
                              hipStream_t stream) {
    const float* query     = (const float*)d_in[0];
    const float* memory    = (const float*)d_in[1];
    const int*   mem_usage = (const int*)  d_in[2];
    const float* in_proj_w = (const float*)d_in[3];
    const float* conv_w    = (const float*)d_in[4];
    const float* conv_b    = (const float*)d_in[5];
    const float* x_proj_w  = (const float*)d_in[6];
    const float* dt_proj_w = (const float*)d_in[7];
    const float* dt_proj_b = (const float*)d_in[8];
    const float* A_log     = (const float*)d_in[9];
    const float* Dvec      = (const float*)d_in[10];
    const float* out_proj_w= (const float*)d_in[11];
    const float* ln_g      = (const float*)d_in[12];
    const float* ln_b      = (const float*)d_in[13];

    float* out_ln    = (float*)d_out;                 // first 1024 floats
    int*   out_usage = ((int*)d_out) + D_MODEL;       // next 131072 ints

    char* ws = (char*)d_ws; size_t off = 0;
    auto alloc = [&](size_t bytes) { void* p = ws + off; off = (off + bytes + 255) & ~(size_t)255; return p; };
    float*          sims      = (float*)         alloc(MEM_SIZE * 4);
    float*          qn        = (float*)         alloc(4);
    int*            idx       = (int*)           alloc(TOPK * 4);
    unsigned short* ret_bf    = (unsigned short*)alloc((size_t)TOPK * D_MODEL * 2);
    unsigned short* w_in_bf   = (unsigned short*)alloc((size_t)2 * D_INNER * D_MODEL * 2);
    unsigned short* w_xp_bf   = (unsigned short*)alloc((size_t)XDBL * D_INNER * 2);
    unsigned short* w_dt_bf   = (unsigned short*)alloc((size_t)D_INNER * DT_RANK * 2);
    float*          xz        = (float*)         alloc((size_t)L_SEQ * 2 * D_INNER * 4);
    float*          u_f32     = (float*)         alloc((size_t)L_SEQ * D_INNER * 4);
    unsigned short* u_bf      = (unsigned short*)alloc((size_t)L_SEQ * D_INNER * 2);
    float*          x_dbl     = (float*)         alloc((size_t)L_SEQ * XDBL * 4);
    unsigned short* dt_bf     = (unsigned short*)alloc((size_t)L_SEQ * DT_RANK * 2);
    float*          deltapre  = (float*)         alloc((size_t)L_SEQ * D_INNER * 4);
    float*          yfinal    = (float*)         alloc(D_INNER * 4);
    float*          ctx       = (float*)         alloc(D_MODEL * 4);
    (void)ws_size; (void)n_in; (void)in_sizes; (void)out_size;

    // retrieval
    qnorm_kernel<<<1, 256, 0, stream>>>(query, qn);
    sims_kernel<<<MEM_SIZE / 8, 256, 0, stream>>>(memory, query, qn, sims);
    copy_usage_kernel<<<MEM_SIZE / 256, 256, 0, stream>>>(mem_usage, out_usage);
    topk_kernel<<<1, 1024, 0, stream>>>(sims, idx);
    gather_kernel<<<TOPK, 256, 0, stream>>>(memory, idx, ret_bf, out_usage);

    // weight converts (f32 -> bf16)
    {
        int n1 = 2 * D_INNER * D_MODEL;
        cvt_bf16_kernel<<<(n1 + 255) / 256, 256, 0, stream>>>(in_proj_w, w_in_bf, n1);
        int n2 = XDBL * D_INNER;
        cvt_bf16_kernel<<<(n2 + 255) / 256, 256, 0, stream>>>(x_proj_w, w_xp_bf, n2);
        int n3 = D_INNER * DT_RANK;
        cvt_bf16_kernel<<<(n3 + 255) / 256, 256, 0, stream>>>(dt_proj_w, w_dt_bf, n3);
    }

    // in_proj (TDM-staged WMMA): xz[64,4096] = retrieved[64,1024] x in_proj_w[4096,1024]^T
    gemm_bf16_tdm_kernel<<<dim3(2 * D_INNER / 16, L_SEQ / 16), 32, 0, stream>>>(
        ret_bf, D_MODEL, w_in_bf, D_MODEL, xz, 2 * D_INNER, D_MODEL);

    // conv + silu
    conv_silu_kernel<<<(L_SEQ * D_INNER) / 256, 256, 0, stream>>>(xz, conv_w, conv_b, u_f32, u_bf);

    // x_proj (direct WMMA): x_dbl[64,96] = u[64,2048] x x_proj_w[96,2048]^T
    gemm_bf16_kernel<<<dim3(XDBL / 16, L_SEQ / 16), 32, 0, stream>>>(
        u_bf, D_INNER, w_xp_bf, D_INNER, x_dbl, XDBL, D_INNER);

    // dt slice -> bf16
    dt_cvt_kernel<<<(L_SEQ * DT_RANK) / 256, 256, 0, stream>>>(x_dbl, dt_bf);

    // dt_proj (direct WMMA): deltapre[64,2048] = dt[64,64] x dt_proj_w[2048,64]^T
    gemm_bf16_kernel<<<dim3(D_INNER / 16, L_SEQ / 16), 32, 0, stream>>>(
        dt_bf, DT_RANK, w_dt_bf, DT_RANK, deltapre, D_INNER, DT_RANK);

    // selective scan (only final y needed) + gating
    scan_kernel<<<D_INNER / 256, 256, 0, stream>>>(deltapre, dt_proj_b, x_dbl, u_f32,
                                                   A_log, Dvec, xz, yfinal);

    // out_proj GEMV + layernorm
    outproj_gemv_kernel<<<D_MODEL / 8, 256, 0, stream>>>(out_proj_w, yfinal, ctx);
    layernorm_kernel<<<1, 1024, 0, stream>>>(ctx, ln_g, ln_b, out_ln);
}
